// Attention1D_32624571580925
// MI455X (gfx1250) — matrix operand order repaired and checked by reading, not compiled, not run on hardware
//
#include <hip/hip_runtime.h>
#include <cstdint>

typedef __attribute__((ext_vector_type(16))) _Float16 v16h;
typedef __attribute__((ext_vector_type(8)))  _Float16 v8h;
typedef __attribute__((ext_vector_type(8)))  float    v8f;

union V16U { v16h v; v8h h[2]; };

// A-fragment (16x32 f16): lane m=l&15, half hh=l>>4.
// elements 0..15 -> K = 16*hh + e (contiguous at p)
__device__ __forceinline__ v16h load_a(const _Float16* p) {
    V16U u;
    u.h[0] = *(const v8h*)(p);
    u.h[1] = *(const v8h*)(p + 8);
    return u.v;
}
// B-fragment (32x16 f16): lane n=l&15, K = 16*hh + e -> 16 contiguous halfs
__device__ __forceinline__ v16h load_b(const _Float16* p) {
    V16U u;
    u.h[0] = *(const v8h*)(p);
    u.h[1] = *(const v8h*)(p + 8);
    return u.v;
}
__device__ __forceinline__ v8f wmma(v16h a, v16h b, v8f c) {
    return __builtin_amdgcn_wmma_f32_16x16x32_f16(false, a, false, b, (short)0, c, false, false);
}
__device__ __forceinline__ uint32_t pk2(float a, float b) {
    union { _Float16 h[2]; uint32_t u; } x;
    x.h[0] = (_Float16)a; x.h[1] = (_Float16)b; return x.u;
}

#define BN 4
#define CC 512
#define TT 2048
#define HH 8
#define DD 64

// ---------------- weight fp32 -> f16 ----------------
__global__ void cvt_w_kernel(const float* __restrict__ qkv_w, const float* __restrict__ proj_w,
                             _Float16* __restrict__ Wq, _Float16* __restrict__ Wp) {
    int i = blockIdx.x * blockDim.x + threadIdx.x;
    if (i < 3 * CC * CC) Wq[i] = (_Float16)qkv_w[i];
    if (i < CC * CC)     Wp[i] = (_Float16)proj_w[i];
}

// ---------------- x [B,C,T] fp32 -> Xt [B,T,C] f16 ----------------
__global__ void xpose_kernel(const float* __restrict__ x, _Float16* __restrict__ Xt) {
    __shared__ float tile[32][33];
    const int b  = blockIdx.z;
    const int t0 = blockIdx.x * 32;
    const int c0 = blockIdx.y * 32;
    const int tx = threadIdx.x, ty = threadIdx.y;
#pragma unroll
    for (int k = 0; k < 32; k += 8)
        tile[ty + k][tx] = x[((size_t)b * CC + c0 + ty + k) * TT + t0 + tx];
    __syncthreads();
#pragma unroll
    for (int k = 0; k < 32; k += 8)
        Xt[((size_t)b * TT + t0 + ty + k) * CC + c0 + tx] = (_Float16)tile[tx][ty + k];
}

// ---------------- QKV GEMM: [1536,512] x [512,2048] per batch ----------------
// block = 128(o) x 64(t) macro-tile (4x2 waves of 32x32) for WGP$ sharing
// outputs: Qt,Kt [B,H,T,64] f16 ; Vm [B,H,64,T] f16   (bias fused)
__global__ void __launch_bounds__(256) qkv_kernel(
    const _Float16* __restrict__ Wq, const float* __restrict__ qkv_b,
    const _Float16* __restrict__ Xt,
    _Float16* __restrict__ Qt, _Float16* __restrict__ Kt, _Float16* __restrict__ Vm) {

    const int w    = threadIdx.x >> 5;
    const int lane = threadIdx.x & 31;
    const int ln = lane & 15, hh = lane >> 4;
    const int b   = blockIdx.x / 384;            // 12 o-macros * 32 t-macros
    const int rem = blockIdx.x % 384;
    const int ot  = (rem >> 5) * 128 + (w >> 1) * 32;
    const int tt  = (rem & 31) * 64  + (w & 1)  * 32;

    v8f D[2][2] = {};
    const _Float16* a0 = Wq + (size_t)(ot + ln) * CC;
    const _Float16* a1 = Wq + (size_t)(ot + 16 + ln) * CC;
    const _Float16* b0 = Xt + ((size_t)b * TT + tt + ln) * CC;
    const _Float16* b1 = Xt + ((size_t)b * TT + tt + 16 + ln) * CC;

    for (int kk = 0; kk < CC; kk += 32) {
        v16h A0 = load_a(a0 + kk + 16 * hh);
        v16h A1 = load_a(a1 + kk + 16 * hh);
        v16h B0 = load_b(b0 + kk + 16 * hh);
        v16h B1 = load_b(b1 + kk + 16 * hh);
        D[0][0] = wmma(A0, B0, D[0][0]);
        D[0][1] = wmma(A0, B1, D[0][1]);
        D[1][0] = wmma(A1, B0, D[1][0]);
        D[1][1] = wmma(A1, B1, D[1][1]);
    }
#pragma unroll
    for (int i = 0; i < 2; i++) {
        const int ob = ot + 16 * i;
        const int p  = ob >> 9;           // 0=q 1=k 2=v (tile never straddles)
        const int h  = (ob & 511) >> 6;
        const int cb = ob & 63;
        float bias[8];
#pragma unroll
        for (int r = 0; r < 8; r++) bias[r] = qkv_b[ob + 8 * hh + r];
#pragma unroll
        for (int j = 0; j < 2; j++) {
            const int t = tt + 16 * j + ln;
            if (p < 2) {
                _Float16* dst = (p == 0 ? Qt : Kt)
                    + ((size_t)(b * HH + h) * TT + t) * DD + cb + 8 * hh;
                v8h o;
#pragma unroll
                for (int r = 0; r < 8; r++) o[r] = (_Float16)(D[i][j][r] + bias[r]);
                *(v8h*)dst = o;
            } else {
#pragma unroll
                for (int r = 0; r < 8; r++)
                    Vm[((size_t)(b * HH + h) * DD + cb + 8 * hh + r) * TT + t] =
                        (_Float16)(D[i][j][r] + bias[r]);
            }
        }
    }
}

// ---------------- flash attention: one wave per 32 queries ----------------
// all 8 waves of a block share one (b,h): K/V line reuse through WGP$
// scores D[s][t] = K(s x c) * Q(c x t) ; AV D[c][t] = V(c x s) * P^T(s x t)
__global__ void __launch_bounds__(256) attn_kernel(
    const _Float16* __restrict__ Qt, const _Float16* __restrict__ Kt,
    const _Float16* __restrict__ Vm, _Float16* __restrict__ AOt) {

    __shared__ __align__(16) _Float16 Pb[8][2][16][32];   // [wave][ttile][t][s] = 16KB
    const int w    = threadIdx.x >> 5;
    const int lane = threadIdx.x & 31;
    const int ln = lane & 15, hh = lane >> 4;
    const int bhid = blockIdx.x >> 3;                 // (b*8+h), 8 blocks each
    const int b = bhid >> 3, h = bhid & 7;
    const int tq = ((blockIdx.x & 7) * 8 + w) * 32;   // 32 queries per wave
    const size_t bh = (size_t)bhid;
    const float scale = 0.044194173824159216f;        // 512^-0.5

    v16h Bq[2][2];                                    // [ttile][kfrag]
#pragma unroll
    for (int tt2 = 0; tt2 < 2; tt2++) {
        const _Float16* qrow = Qt + (bh * TT + tq + 16 * tt2 + ln) * DD + 16 * hh;
        Bq[tt2][0] = load_b(qrow);
        Bq[tt2][1] = load_b(qrow + 32);
    }

    float m[2] = { -3.0e38f, -3.0e38f }, lsum[2] = { 0.f, 0.f };
    v8f O[2][4] = {};                                 // [ttile][ct]

    for (int s0 = 0; s0 < TT; s0 += 32) {
        v8f S[2][2];                                  // [ttile][tau]
#pragma unroll
        for (int tau = 0; tau < 2; tau++) {
            const _Float16* krow = Kt + (bh * TT + s0 + 16 * tau + ln) * DD + 16 * hh;
            v16h Ak0 = load_a(krow);
            v16h Ak1 = load_a(krow + 32);
#pragma unroll
            for (int tt2 = 0; tt2 < 2; tt2++) {
                v8f acc = {};
                acc = wmma(Ak0, Bq[tt2][0], acc);
                acc = wmma(Ak1, Bq[tt2][1], acc);
                S[tt2][tau] = acc;
            }
        }
#pragma unroll
        for (int tt2 = 0; tt2 < 2; tt2++) {
            float mb = -3.0e38f;
#pragma unroll
            for (int tau = 0; tau < 2; tau++)
#pragma unroll
                for (int r = 0; r < 8; r++) { S[tt2][tau][r] *= scale; mb = fmaxf(mb, S[tt2][tau][r]); }
            mb = fmaxf(mb, __shfl_xor(mb, 16, 32));
            const float mnew = fmaxf(m[tt2], mb);
            const float f = __expf(m[tt2] - mnew);
            float P[2][8];
            float ls = 0.f;
#pragma unroll
            for (int tau = 0; tau < 2; tau++)
#pragma unroll
                for (int r = 0; r < 8; r++) { P[tau][r] = __expf(S[tt2][tau][r] - mnew); ls += P[tau][r]; }
            ls += __shfl_xor(ls, 16, 32);
            lsum[tt2] = lsum[tt2] * f + ls;
            m[tt2] = mnew;
#pragma unroll
            for (int ct = 0; ct < 4; ct++)
#pragma unroll
                for (int r = 0; r < 8; r++) O[tt2][ct][r] *= f;

            // P (D-layout) -> LDS [t][s], reload as B-fragment (s contiguous)
            _Float16* pb = &Pb[w][tt2][ln][0];
#pragma unroll
            for (int tau = 0; tau < 2; tau++)
#pragma unroll
                for (int r = 0; r < 8; r += 2)
                    *(uint32_t*)&pb[16 * tau + 8 * hh + r] = pk2(P[tau][r], P[tau][r + 1]);
        }
        const v16h Bp0 = load_b(&Pb[w][0][ln][16 * hh]);
        const v16h Bp1 = load_b(&Pb[w][1][ln][16 * hh]);
#pragma unroll
        for (int ct = 0; ct < 4; ct++) {
            const _Float16* vrow = Vm + (bh * DD + ct * 16 + ln) * TT + s0 + 16 * hh;
            const v16h Av = load_a(vrow);
            O[0][ct] = wmma(Av, Bp0, O[0][ct]);
            O[1][ct] = wmma(Av, Bp1, O[1][ct]);
        }
    }
#pragma unroll
    for (int tt2 = 0; tt2 < 2; tt2++) {
        const float inv = 1.0f / lsum[tt2];
        _Float16* ao = AOt + ((size_t)b * TT + tq + 16 * tt2 + ln) * CC + h * DD + 8 * hh;
#pragma unroll
        for (int ct = 0; ct < 4; ct++) {
            v8h o;
#pragma unroll
            for (int r = 0; r < 8; r++) o[r] = (_Float16)(O[tt2][ct][r] * inv);
            *(v8h*)(ao + ct * 16) = o;
        }
    }
}

// ---------------- proj GEMM + bias + residual ----------------
__global__ void __launch_bounds__(256) proj_kernel(
    const _Float16* __restrict__ Wp, const float* __restrict__ proj_b,
    const _Float16* __restrict__ AOt, const float* __restrict__ x,
    float* __restrict__ out) {

    const int w    = threadIdx.x >> 5;
    const int lane = threadIdx.x & 31;
    const int ln = lane & 15, hh = lane >> 4;
    const int b   = blockIdx.x / 128;            // 4 o-macros * 32 t-macros
    const int rem = blockIdx.x % 128;
    const int ot  = (rem >> 5) * 128 + (w >> 1) * 32;
    const int tt  = (rem & 31) * 64  + (w & 1)  * 32;

    v8f D[2][2] = {};
    const _Float16* a0 = Wp + (size_t)(ot + ln) * CC;
    const _Float16* a1 = Wp + (size_t)(ot + 16 + ln) * CC;
    const _Float16* b0 = AOt + ((size_t)b * TT + tt + ln) * CC;
    const _Float16* b1 = AOt + ((size_t)b * TT + tt + 16 + ln) * CC;

    for (int kk = 0; kk < CC; kk += 32) {
        v16h A0 = load_a(a0 + kk + 16 * hh);
        v16h A1 = load_a(a1 + kk + 16 * hh);
        v16h B0 = load_b(b0 + kk + 16 * hh);
        v16h B1 = load_b(b1 + kk + 16 * hh);
        D[0][0] = wmma(A0, B0, D[0][0]);
        D[0][1] = wmma(A0, B1, D[0][1]);
        D[1][0] = wmma(A1, B0, D[1][0]);
        D[1][1] = wmma(A1, B1, D[1][1]);
    }
#pragma unroll
    for (int i = 0; i < 2; i++) {
        const int ob = ot + 16 * i + 8 * hh;
#pragma unroll
        for (int j = 0; j < 2; j++) {
            const int t = tt + 16 * j + ln;
#pragma unroll
            for (int r = 0; r < 8; r++) {
                const int o = ob + r;
                const size_t idx = ((size_t)b * CC + o) * TT + t;
                out[idx] = x[idx] + D[i][j][r] + proj_b[o];
            }
        }
    }
}

extern "C" void kernel_launch(void* const* d_in, const int* in_sizes, int n_in,
                              void* d_out, int out_size, void* d_ws, size_t ws_size,
                              hipStream_t stream) {
    const float* x      = (const float*)d_in[0];
    const float* qkv_w  = (const float*)d_in[1];
    const float* qkv_b  = (const float*)d_in[2];
    const float* proj_w = (const float*)d_in[3];
    const float* proj_b = (const float*)d_in[4];
    float* out = (float*)d_out;

    char* ws = (char*)d_ws;
    _Float16* Wq  = (_Float16*)(ws);                 // 1536*512*2   = 1.5 MB
    _Float16* Wp  = (_Float16*)(ws + 1572864);       // 512*512*2    = 0.5 MB
    _Float16* Xt  = (_Float16*)(ws + 2097152);       // [B,T,C] 8 MB
    _Float16* Qt  = (_Float16*)(ws + 10485760);      // [B,H,T,64] 8 MB
    _Float16* Kt  = (_Float16*)(ws + 18874368);      // [B,H,T,64] 8 MB
    _Float16* Vm  = (_Float16*)(ws + 27262976);      // [B,H,64,T] 8 MB
    _Float16* AOt = (_Float16*)(ws + 35651584);      // [B,T,C] 8 MB

    cvt_w_kernel<<<3072, 256, 0, stream>>>(qkv_w, proj_w, Wq, Wp);
    xpose_kernel<<<dim3(TT / 32, CC / 32, BN), dim3(32, 8), 0, stream>>>(x, Xt);
    qkv_kernel<<<1536, 256, 0, stream>>>(Wq, qkv_b, Xt, Qt, Kt, Vm);   // 12288 waves
    attn_kernel<<<256, 256, 0, stream>>>(Qt, Kt, Vm, AOt);             // 2048 waves, 32q each
    proj_kernel<<<512, 256, 0, stream>>>(Wp, proj_b, AOt, x, out);     // 4096 waves
}
